// InstanceAggLayer_58815282152043
// MI455X (gfx1250) — matrix operand order repaired and checked by reading, not compiled, not run on hardware
//
#include <hip/hip_runtime.h>
#include <hip/hip_bf16.h>

// ---------------------------------------------------------------------------
// GDCD instance aggregation, MI455X (gfx1250, wave32, WMMA + TDM).
//
//   f_x = feats_x @ P_x                      (WMMA f16 -> f32, stored f16)
//   out = leaky_relu(gather4(f_*) @ W_GDCD)  (fused gather + WMMA edge GEMM)
//
// Weights are pre-swizzled into WMMA-fragment order so every B-fragment read
// is one contiguous 1KB block (32 lanes x 32B). The edge kernel stages the
// 128KB swizzled W in LDS via the Tensor Data Mover (fallback: cooperative
// copy), and each wave computes TWO 16-edge M-tiles so each LDS B fragment
// feeds two WMMAs.
//
// Workspace layout (bytes):
//   [0,       128K)  : Wswz f16, fragment-swizzled W_GDCD   (16 ksteps x 8 n)
//   [128K,    224K)  : Pswz f16, fragment-swizzled P_x x3   ( 4 ksteps x 8 n)
//   [256K, 256K+77M) : f_disease/f_gene/f_chemical f16 [100000][128]
// ---------------------------------------------------------------------------

typedef __attribute__((ext_vector_type(16))) _Float16 v16h;
typedef __attribute__((ext_vector_type(8)))  _Float16 v8h;
typedef __attribute__((ext_vector_type(8)))  float    v8f;
typedef __attribute__((ext_vector_type(4)))  unsigned int u32x4;
typedef __attribute__((ext_vector_type(8)))  int          i32x8;
typedef __attribute__((ext_vector_type(4)))  int          i32x4;

#define N_NODES 100000
#define IN_DIM  128
#define OUT_DIM 128
#define N_EDGES 1000000
#define ALPHA   0.2f
#define W_BYTES (512 * 128 * 2)   // 128 KB swizzled W tile

#if __has_builtin(__builtin_amdgcn_tensor_load_to_lds) && \
    __has_builtin(__builtin_amdgcn_s_wait_tensorcnt)
#define HAVE_TDM 1
#else
#define HAVE_TDM 0
#endif

static __device__ __forceinline__ v16h cat8(v8h lo, v8h hi) {
    return __builtin_shufflevector(lo, hi,
        0, 1, 2, 3, 4, 5, 6, 7, 8, 9, 10, 11, 12, 13, 14, 15);
}

static __device__ __forceinline__ v16h frag_b(const _Float16* base, int frag, int lane) {
    const _Float16* bp = base + (((size_t)frag * 32 + lane) << 4);  // 32B/lane
    return cat8(*(const v8h*)bp, *(const v8h*)(bp + 8));
}

// ---------------------------------------------------------------------------
// Swizzle f32 W[K][N] (row-major) into f16 WMMA B-fragment order:
//   out[((ks*(N/16)+n)*32 + L)*16 + j] = W[ks*32 + (L>>4)*16 + j][n*16 + (L&15)]
// Fragment (ks,n) is a contiguous 1KB block; lane L owns 32 contiguous bytes.
// ---------------------------------------------------------------------------
__global__ void swizzle_w_kernel(const float* __restrict__ in,
                                 _Float16* __restrict__ out, int K, int N) {
    int idx = blockIdx.x * blockDim.x + threadIdx.x;
    if (idx >= K * N) return;
    int f   = idx >> 9;          // fragment id
    int rem = idx & 511;
    int L   = rem >> 4;          // lane
    int j   = rem & 15;
    int NT  = N >> 4;
    int ks  = f / NT;
    int n   = f - ks * NT;
    int k   = ks * 32 + ((L >> 4) << 4) + j;
    int col = n * 16 + (L & 15);
    out[idx] = (_Float16)in[(size_t)k * N + col];
}

// ---------------------------------------------------------------------------
// Node GEMM: f16 f[100000][128] = f32 feats @ P   (Pswz fragment-swizzled)
// ---------------------------------------------------------------------------
__global__ __launch_bounds__(256)
void node_gemm_kernel(const float* __restrict__ feats0,
                      const float* __restrict__ feats1,
                      const float* __restrict__ feats2,
                      const _Float16* __restrict__ PswzBase,  // 3 * [4*8 frags]
                      _Float16* __restrict__ fBase) {
    const int lane = threadIdx.x & 31;
    const int wave = blockIdx.x * (blockDim.x >> 5) + (threadIdx.x >> 5);
    const int MT = N_NODES / 16;          // 6250
    if (wave >= 3 * MT) return;
    const int t = wave / MT;
    const int mtile = wave - t * MT;

    const float* feats = (t == 0) ? feats0 : (t == 1) ? feats1 : feats2;
    const _Float16* Pswz = PswzBase + (size_t)t * IN_DIM * OUT_DIM;
    _Float16* fout = fBase + (size_t)t * N_NODES * OUT_DIM;

    const int m = lane & 15;
    const int g = (lane >> 4) & 1;
    const float* arow = feats + (size_t)(mtile * 16 + m) * IN_DIM;

    v8f zero = {};
    v8f c[8];
#pragma unroll
    for (int n = 0; n < 8; ++n) c[n] = zero;

#pragma unroll
    for (int ks = 0; ks < 4; ++ks) {
        const float* p0 = arow + ks * 32 + g * 16;
        const float* p1 = p0 + 8;
        v8h lo, hi;
#pragma unroll
        for (int i = 0; i < 8; ++i) {
            lo[i] = (_Float16)p0[i];
            hi[i] = (_Float16)p1[i];
        }
        v16h a = cat8(lo, hi);
#pragma unroll
        for (int n = 0; n < 8; ++n) {
            v16h b = frag_b(Pswz, ks * 8 + n, lane);
            c[n] = __builtin_amdgcn_wmma_f32_16x16x32_f16(
                false, a, false, b, (short)0, c[n], false, false);
        }
    }

#pragma unroll
    for (int n = 0; n < 8; ++n)
#pragma unroll
        for (int r = 0; r < 8; ++r)
            fout[(size_t)(mtile * 16 + g * 8 + r) * OUT_DIM + n * 16 + m] =
                (_Float16)c[n][r];
}

// ---------------------------------------------------------------------------
// Fused gather + edge GEMM + LeakyReLU. One wave = two 16-edge M-tiles.
// Swizzled W staged in LDS (TDM if available); B fragments via ds_load_b128.
// ---------------------------------------------------------------------------
__global__ __launch_bounds__(256)
void edge_gemm_kernel(const int* __restrict__ adj,
                      const _Float16* __restrict__ fdis,
                      const _Float16* __restrict__ fgene,
                      const _Float16* __restrict__ fchem,
                      const _Float16* __restrict__ Wswz,   // 128KB swizzled
                      float* __restrict__ out) {
    extern __shared__ __attribute__((aligned(16))) _Float16 smem[];

    // ---- stage swizzled W into LDS (whole block cooperates) ----
#if HAVE_TDM
    if (threadIdx.x < 32) {
        // Tensor DMA descriptor: 1-D tile of 16384 x 8B elements (= 128KB).
        const unsigned lds_off = (unsigned)(size_t)(void*)smem;  // low32 = LDS byte offset
        const unsigned long long ga = (unsigned long long)(size_t)Wswz;
        u32x4 g0 = {};
        i32x8 g1 = {};
        i32x4 g2 = {};
        i32x4 g3 = {};
        i32x8 g4 = {};
        g0[0] = 1u;                                           // count = 1
        g0[1] = lds_off;                                      // lds_addr
        g0[2] = (unsigned)ga;                                 // global_addr[31:0]
        g0[3] = ((unsigned)(ga >> 32) & 0x01FFFFFFu)          // global_addr[56:32]
              | (2u << 30);                                   // type = 2 ("image")
        g1[0] = (int)(3u << 16);        // workgroup_mask=0, data_size=3 (8B)
        g1[1] = (int)(16384u << 16);    // tensor_dim0[15:0] in bits 63:48
        g1[2] = (int)(1u << 16);        // tensor_dim0 hi=0, tensor_dim1=1
        g1[3] = (int)(16384u << 16);    // tile_dim0 = 16384
        g1[4] = 1;                      // tile_dim1 = 1, tile_dim2 = 0
        g1[5] = 16384;                  // tensor_dim0_stride[31:0]
        g1[6] = (int)(16384u << 16);    // stride0 hi=0, tensor_dim1_stride[15:0]
        g1[7] = 0;
        __builtin_amdgcn_tensor_load_to_lds(g0, g1, g2, g3, g4, 0);
        __builtin_amdgcn_s_wait_tensorcnt(0);
    }
    __syncthreads();
#else
    {
        const v8h* src = (const v8h*)Wswz;
        v8h* dst = (v8h*)smem;
        for (int i = threadIdx.x; i < W_BYTES / 16; i += blockDim.x)
            dst[i] = src[i];
    }
    __syncthreads();
#endif

    const int lane = threadIdx.x & 31;
    const int wave = blockIdx.x * (blockDim.x >> 5) + (threadIdx.x >> 5);
    const int m = lane & 15;
    const int g = (lane >> 4) & 1;
    const int PAIRS = N_EDGES / 32;      // 31250 wave-tasks, 2 M-tiles each

    if (wave < PAIRS) {
        const int eb0 = wave * 32;       // tiles [eb0, eb0+16)
        const int eb1 = eb0 + 16;

        const _Float16* rpA[4];
        const _Float16* rpB[4];
        rpA[0] = fdis  + (size_t)adj[0 * N_EDGES + eb0 + m] * OUT_DIM;
        rpA[1] = fdis  + (size_t)adj[1 * N_EDGES + eb0 + m] * OUT_DIM;
        rpA[2] = fgene + (size_t)adj[2 * N_EDGES + eb0 + m] * OUT_DIM;
        rpA[3] = fchem + (size_t)adj[3 * N_EDGES + eb0 + m] * OUT_DIM;
        rpB[0] = fdis  + (size_t)adj[0 * N_EDGES + eb1 + m] * OUT_DIM;
        rpB[1] = fdis  + (size_t)adj[1 * N_EDGES + eb1 + m] * OUT_DIM;
        rpB[2] = fgene + (size_t)adj[2 * N_EDGES + eb1 + m] * OUT_DIM;
        rpB[3] = fchem + (size_t)adj[3 * N_EDGES + eb1 + m] * OUT_DIM;

        v8f zero = {};
        v8f c0[8], c1[8];
#pragma unroll
        for (int n = 0; n < 8; ++n) { c0[n] = zero; c1[n] = zero; }

#pragma unroll
        for (int seg = 0; seg < 4; ++seg) {
#pragma unroll
            for (int kk = 0; kk < 4; ++kk) {
                const int kstep = seg * 4 + kk;
                const _Float16* apA = rpA[seg] + kk * 32 + g * 16;
                const _Float16* apB = rpB[seg] + kk * 32 + g * 16;
                v16h a0 = cat8(*(const v8h*)apA, *(const v8h*)(apA + 8));
                v16h a1 = cat8(*(const v8h*)apB, *(const v8h*)(apB + 8));
#pragma unroll
                for (int n = 0; n < 8; ++n) {
                    v16h b = frag_b(smem, kstep * 8 + n, lane);  // ds_load_b128 x2
                    c0[n] = __builtin_amdgcn_wmma_f32_16x16x32_f16(
                        false, a0, false, b, (short)0, c0[n], false, false);
                    c1[n] = __builtin_amdgcn_wmma_f32_16x16x32_f16(
                        false, a1, false, b, (short)0, c1[n], false, false);
                }
            }
        }

        // LeakyReLU + fp32 store; each half-wave row chunk is 64B contiguous.
#pragma unroll
        for (int n = 0; n < 8; ++n)
#pragma unroll
            for (int r = 0; r < 8; ++r) {
                float v0 = c0[n][r];
                float v1 = c1[n][r];
                v0 = (v0 > 0.f) ? v0 : ALPHA * v0;
                v1 = (v1 > 0.f) ? v1 : ALPHA * v1;
                out[(size_t)(eb0 + g * 8 + r) * OUT_DIM + n * 16 + m] = v0;
                out[(size_t)(eb1 + g * 8 + r) * OUT_DIM + n * 16 + m] = v1;
            }
    }
}

extern "C" void kernel_launch(void* const* d_in, const int* in_sizes, int n_in,
                              void* d_out, int out_size, void* d_ws, size_t ws_size,
                              hipStream_t stream) {
    const float* dis  = (const float*)d_in[0];
    const float* gen  = (const float*)d_in[1];
    const float* chem = (const float*)d_in[2];
    // d_in[3] (species_feats) and d_in[8] (P_species) are dead for GDCD output.
    const int*   adj  = (const int*)d_in[4];
    const float* Pd   = (const float*)d_in[5];
    const float* Pg   = (const float*)d_in[6];
    const float* Pc   = (const float*)d_in[7];
    const float* W    = (const float*)d_in[9];
    float* out = (float*)d_out;

    char* ws = (char*)d_ws;
    _Float16* Wswz = (_Float16*)ws;                     // 128 KB
    _Float16* Pswz = (_Float16*)(ws + (128 << 10));     // 3 * 32 KB
    _Float16* fB   = (_Float16*)(ws + (256 << 10));     // 3 * 25.6 MB
    const size_t fStride = (size_t)N_NODES * OUT_DIM;

    // Weight swizzle + f32->f16 (tiny).
    swizzle_w_kernel<<<(512 * 128 + 255) / 256, 256, 0, stream>>>(W, Wswz, 512, 128);
    swizzle_w_kernel<<<(128 * 128 + 255) / 256, 256, 0, stream>>>(Pd, Pswz + 0 * 16384, 128, 128);
    swizzle_w_kernel<<<(128 * 128 + 255) / 256, 256, 0, stream>>>(Pg, Pswz + 1 * 16384, 128, 128);
    swizzle_w_kernel<<<(128 * 128 + 255) / 256, 256, 0, stream>>>(Pc, Pswz + 2 * 16384, 128, 128);

    // Node GEMMs: 3 * 6250 M-tiles, 8 waves per block.
    {
        int waves = 3 * (N_NODES / 16);
        int blocks = (waves + 7) / 8;
        node_gemm_kernel<<<blocks, 256, 0, stream>>>(dis, gen, chem, Pswz, fB);
    }

    // Fused gather + edge GEMM: 31250 wave-tasks (2 M-tiles each), 128KB LDS.
    {
        (void)hipFuncSetAttribute((const void*)edge_gemm_kernel,
                                  hipFuncAttributeMaxDynamicSharedMemorySize,
                                  W_BYTES);
        int waves = N_EDGES / 32;
        int blocks = (waves + 7) / 8;
        edge_gemm_kernel<<<blocks, 256, W_BYTES, stream>>>(
            adj, fB, fB + fStride, fB + 2 * fStride, Wswz, out);
    }
}
